// SplicePredictorHybrid_20177756357347
// MI455X (gfx1250) — matrix operand order repaired and checked by reading, not compiled, not run on hardware
//
#include <hip/hip_runtime.h>
#include <hip/hip_bf16.h>
#include <math.h>

// ---------------------------------------------------------------------------
// SplicePredictorHybrid forward for MI455X (gfx1250, wave32, WMMA).
//  - all GEMM-shaped ops run on v_wmma_f32_16x16x32_f16
//  - fragments are loaded as contiguous 16B vectors (b128), boundary handling
//    via clamped address + select (no exec-mask branch storms)
//  - transformer linears: 8-wave workgroup tile, LDS double buffering with
//    gfx1250 async global->LDS copies (global_load_async_to_lds_b128)
// ---------------------------------------------------------------------------

typedef __attribute__((ext_vector_type(16))) _Float16 v16h;
typedef __attribute__((ext_vector_type(8)))  _Float16 h8;
typedef __attribute__((ext_vector_type(2)))  _Float16 h2;
typedef __attribute__((ext_vector_type(8)))  float    v8f;
typedef int vint4 __attribute__((vector_size(16)));

#define DEV __device__ __forceinline__
#define AS1 __attribute__((address_space(1)))
#define AS3 __attribute__((address_space(3)))

#if defined(__has_builtin)
#if __has_builtin(__builtin_amdgcn_global_load_async_to_lds_b128)
#define HAVE_ASYNC_CP 1
#endif
#endif

constexpr int BB   = 64;
constexpr int LL   = 504;
constexpr int HID  = 768;
constexpr int DD   = 128;
constexpr int NN   = 128;
constexpr int HH   = 4;
constexpr int KSZ  = 8;
constexpr int LMAX = 512;
constexpr int LC   = 505;          // conv output length
constexpr int LP   = 512;          // padded length
constexpr int HDIM = 32;           // head dim
constexpr int QKV  = 384;
constexpr int FF   = 256;
constexpr int MROWS = LC * BB;     // 32320 = 64 * 505
constexpr int CONVK = KSZ * HID;   // 6144

// f16 weight arena offsets (elements)
constexpr size_t W_CONV = 0;                         // (D, KS*HID) reordered
constexpr size_t W_QKV  = W_CONV + (size_t)DD * CONVK;
constexpr size_t W_OUT  = W_QKV + (size_t)2 * QKV * DD;
constexpr size_t W_FF1  = W_OUT + (size_t)2 * DD * DD;
constexpr size_t W_FF2  = W_FF1 + (size_t)2 * FF * DD;
constexpr size_t W_TOT  = W_FF2 + (size_t)2 * DD * FF;   // 1048576

// ---------------- async global->LDS helpers --------------------------------
DEV void cp16(const _Float16* g, _Float16* l) {
#ifdef HAVE_ASYNC_CP
  __builtin_amdgcn_global_load_async_to_lds_b128(
      (AS1 vint4*)(uintptr_t)g,
      (AS3 vint4*)(unsigned)(uintptr_t)l,
      0, 0);
#else
  *(uint4*)l = *(const uint4*)g;
#endif
}

template <int N> DEV void wait_async() {
#ifdef HAVE_ASYNC_CP
#if __has_builtin(__builtin_amdgcn_s_wait_asynccnt)
  __builtin_amdgcn_s_wait_asynccnt((short)N);
#else
  asm volatile("s_wait_asynccnt %0" ::"i"(N) : "memory");
#endif
#endif
}

// ---------------- WMMA fragment builders (CDNA5 ISA 7.12.2, wave32) --------
// Per-lane data is contiguous:
//   A frag: A[m][hi*16 .. +15]                        (one 32B run)
//   B frag: B[hi*16 .. +15][n]                        (one 32B run)
//   C tile: lane n=lane&15; VGPR r holds row m=(lane>>4)*8+r
DEV v16h combine(h8 x0, h8 x1) {
  v16h a;
#pragma unroll
  for (int i = 0; i < 8; ++i) { a[i] = x0[i]; a[8 + i] = x1[i]; }
  return a;
}
DEV v16h frag_a_ptr(const _Float16* row, int hi, bool valid) {
  h8 z = {};
  h8 x0 = *(const h8*)(row + hi * 16);
  h8 x1 = *(const h8*)(row + hi * 16 + 8);
  return combine(valid ? x0 : z, valid ? x1 : z);
}
DEV v16h frag_b_ptr(const _Float16* col, int hi, bool valid) {
  h8 z = {};
  h8 x0 = *(const h8*)(col + hi * 16);
  h8 x1 = *(const h8*)(col + hi * 16 + 8);
  return combine(valid ? x0 : z, valid ? x1 : z);
}
// generic per-pair B builder (for strided operands, e.g. V in P@V)
template <class F2>
DEV v16h frag_b_gen(F2 ld2, int lo, int hi) {
  v16h b;
#pragma unroll
  for (int j = 0; j < 8; ++j) {
    const int kb = hi * 16 + 2 * j;
    h2 p = ld2(kb, lo);
    b[2 * j] = p[0]; b[2 * j + 1] = p[1];
  }
  return b;
}

DEV v8f wmma_acc(v16h a, v16h b, v8f c) {
  return __builtin_amdgcn_wmma_f32_16x16x32_f16(false, a, false, b, (short)0, c,
                                                false, false);
}

DEV void store_tile(v8f c, float* Cf, _Float16* Ch, long rs, long cs,
                    int m0, int n0, int mBound, int nBound,
                    const float* biasRow, const float* biasCol,
                    bool relu, float alpha) {
  const int lane = threadIdx.x & 31;
  const int lo = lane & 15;
  const int hi = lane >> 4;
  const int n = n0 + lo;
  if (n >= nBound) return;
#pragma unroll
  for (int r = 0; r < 8; ++r) {
    const int m = m0 + hi * 8 + r;
    if (m >= mBound) continue;
    float v = c[r] * alpha;
    if (biasRow) v += biasRow[m];
    if (biasCol) v += biasCol[n];
    if (relu) v = fmaxf(v, 0.f);
    const long o = (long)m * rs + (long)n * cs;
    if (Cf) Cf[o] = v;
    if (Ch) Ch[o] = (_Float16)v;
  }
}

// ---------------------------------------------------------------------------
// 1) masked embedding gather -> f16 (B,L,HID)
__global__ void k_embed(const int* __restrict__ ids, const int* __restrict__ mask,
                        const float* __restrict__ tab, _Float16* __restrict__ embH) {
  size_t idx = (size_t)blockIdx.x * blockDim.x + threadIdx.x;
  const size_t total = (size_t)BB * LL * HID;
  if (idx >= total) return;
  const int h = (int)(idx % HID);
  const size_t bl = idx / HID;
  float v = 0.f;
  if (mask[bl]) v = tab[(size_t)ids[bl] * HID + h];
  embH[idx] = (_Float16)v;
}

// 2) convert all weights to an f16 arena (conv reordered to (D, KS*HID))
__global__ void k_wcvt(const float* __restrict__ conv_w, const float* __restrict__ qkv_w,
                       const float* __restrict__ out_w, const float* __restrict__ ff1_w,
                       const float* __restrict__ ff2_w, _Float16* __restrict__ wH) {
  const size_t idx = (size_t)blockIdx.x * blockDim.x + threadIdx.x;
  if (idx >= W_TOT) return;
  float v;
  if (idx < W_QKV) {
    const size_t d = idx / CONVK, r = idx % CONVK;
    const size_t ks = r / HID, h = r % HID;
    v = conv_w[(d * HID + h) * KSZ + ks];
  } else if (idx < W_OUT) v = qkv_w[idx - W_QKV];
  else if (idx < W_FF1)   v = out_w[idx - W_OUT];
  else if (idx < W_FF2)   v = ff1_w[idx - W_FF1];
  else                    v = ff2_w[idx - W_FF2];
  wH[idx] = (_Float16)v;
}

// 3) conv1d (pad 4, K=8) as implicit GEMM; wave computes 16(d) x 64(l),
//    A fragment (weights) reused across 4 B fragments. Output (b,l,d) f32.
__global__ void k_conv(const _Float16* __restrict__ embH, const _Float16* __restrict__ convH,
                       const float* __restrict__ convb, float* __restrict__ xconv) {
  const int n0 = blockIdx.x * 64;   // l base
  const int m0 = blockIdx.y * 16;   // d base
  const int b  = blockIdx.z;
  const int lane = threadIdx.x & 31, lo = lane & 15, hi = lane >> 4;
  v8f c[4] = {v8f{}, v8f{}, v8f{}, v8f{}};
  const _Float16* arow = convH + (size_t)(m0 + lo) * CONVK;
  for (int kk = 0; kk < CONVK; kk += 32) {
    const int ks = kk / HID, h0 = kk - ks * HID;
    v16h a = frag_a_ptr(arow + kk, hi, true);
#pragma unroll
    for (int ns = 0; ns < 4; ++ns) {
      const int pos = n0 + ns * 16 + lo + ks - 4;
      const bool ok = (pos >= 0) & (pos < LL);
      const int pc = pos < 0 ? 0 : (pos >= LL ? LL - 1 : pos);
      v16h bf = frag_b_ptr(embH + ((size_t)b * LL + pc) * HID + h0, hi, ok);
      c[ns] = wmma_acc(a, bf, c[ns]);
    }
  }
#pragma unroll
  for (int ns = 0; ns < 4; ++ns)
    store_tile(c[ns], xconv + (size_t)b * LP * DD, nullptr, /*rs=*/1, /*cs=*/DD,
               m0, n0 + ns * 16, DD, LP, convb, nullptr, true, 1.f);
}

// 4a) S4D discretization per (d,n)
__global__ void k_s4cd(const float* __restrict__ log_dt, const float* __restrict__ logAre,
                       const float* __restrict__ Aim, const float* __restrict__ Cre,
                       const float* __restrict__ Cim, float* __restrict__ cd,
                       float* __restrict__ dtA) {
  const int idx = blockIdx.x * blockDim.x + threadIdx.x;
  if (idx >= DD * NN) return;
  const int d = idx / NN;
  const float dt = expf(log_dt[d]);
  const float ar = -expf(logAre[idx]);
  const float ai = Aim[idx];
  const float x = ar * dt, y = ai * dt;
  const float ex = expf(x);
  const float er = ex * cosf(y) - 1.f;
  const float ei = ex * sinf(y);
  const float cr = Cre[idx], ci = Cim[idx];
  const float nr = cr * er - ci * ei;
  const float ni = cr * ei + ci * er;
  const float den = ar * ar + ai * ai;
  cd[2 * idx]     = (nr * ar + ni * ai) / den;
  cd[2 * idx + 1] = (ni * ar - nr * ai) / den;
  dtA[2 * idx]     = x;
  dtA[2 * idx + 1] = y;
}

// 4b) S4D kernel Kc[l,d] = 2*Re sum_n Cd * exp(dtA*l)
__global__ void k_s4k(const float* __restrict__ cd, const float* __restrict__ dtA,
                      float* __restrict__ Kc) {
  const int idx = blockIdx.x * blockDim.x + threadIdx.x;
  if (idx >= LMAX * DD) return;
  const int d = idx % DD;
  const int l = idx / DD;
  float acc = 0.f;
  for (int n = 0; n < NN; ++n) {
    const int q = d * NN + n;
    const float x = dtA[2 * q], y = dtA[2 * q + 1];
    const float el = expf(x * (float)l);
    acc += el * (cd[2 * q] * cosf(y * (float)l) - cd[2 * q + 1] * sinf(y * (float)l));
  }
  Kc[(size_t)l * DD + d] = 2.f * acc;
}

// 5) direct causal conv (== causal part of the FFT conv) + skip; relayout to
//    transformer order, emit f32 master + f16 mirror.
__global__ void k_s4conv(const float* __restrict__ xconv, const float* __restrict__ Kc,
                         const float* __restrict__ s4d, float* __restrict__ xF,
                         _Float16* __restrict__ xH) {
  size_t idx = (size_t)blockIdx.x * blockDim.x + threadIdx.x;
  if (idx >= (size_t)DD * BB * LC) return;
  const int d = (int)(idx % DD);
  const size_t r = idx / DD;
  const int b = (int)(r % BB);
  const int l = (int)(r / BB);
  const float* xc = xconv + (size_t)b * LP * DD + d;
  float acc = s4d[d] * xc[(size_t)l * DD];
  for (int m = 0; m <= l; ++m)
    acc += Kc[(size_t)(l - m) * DD + d] * xc[(size_t)m * DD];
  const size_t o = ((size_t)l * BB + b) * DD + d;
  xF[o] = acc;
  xH[o] = (_Float16)acc;
}

// 6) tiled linear: C[M x N] = A[M x K] @ W[N x K]^T (+bias, opt relu)
//    64x128 workgroup tile, 8 waves, BK=32, LDS double-buffered, async copies.
__global__ void __launch_bounds__(256)
k_gemm_tiled(const _Float16* __restrict__ A, const _Float16* __restrict__ W,
             const float* __restrict__ bias, float* __restrict__ Cf,
             _Float16* __restrict__ Ch, int ldc, int K, int relu) {
  alignas(16) __shared__ _Float16 As[2][64 * 32];
  alignas(16) __shared__ _Float16 Bs[2][128 * 32];
  const int m0 = blockIdx.x * 64;
  const int n0 = blockIdx.y * 128;
  const int t = threadIdx.x;
  const int lane = t & 31, lo = lane & 15, hi = lane >> 4;
  const int wid = t >> 5, wm = wid & 3, wn = wid >> 2;

  auto stage = [&](int buf, int kk) {
    { const int row = t >> 2, seg = t & 3;
      cp16(A + (size_t)(m0 + row) * K + kk + seg * 8, &As[buf][row * 32 + seg * 8]); }
#pragma unroll
    for (int q = 0; q < 2; ++q) {
      const int idx = t + q * 256, row = idx >> 2, seg = idx & 3;
      cp16(W + (size_t)(n0 + row) * K + kk + seg * 8, &Bs[buf][row * 32 + seg * 8]);
    }
  };

  v8f c[4] = {v8f{}, v8f{}, v8f{}, v8f{}};
  const int nk = K / 32;
  stage(0, 0);
  for (int ic = 0; ic < nk; ++ic) {
    const int cur = ic & 1;
    if (ic + 1 < nk) { stage(cur ^ 1, (ic + 1) * 32); wait_async<3>(); }
    else             { wait_async<0>(); }
    __syncthreads();
    const _Float16* As_ = &As[cur][0];
    const _Float16* Bs_ = &Bs[cur][0];
    v16h a = frag_a_ptr(As_ + (wm * 16 + lo) * 32, hi, true);
#pragma unroll
    for (int ns = 0; ns < 4; ++ns) {
      v16h bf = frag_b_ptr(Bs_ + (wn * 64 + ns * 16 + lo) * 32, hi, true);
      c[ns] = wmma_acc(a, bf, c[ns]);
    }
    __syncthreads();
  }
#pragma unroll
  for (int ns = 0; ns < 4; ++ns)
    store_tile(c[ns], Cf, Ch, ldc, 1, m0 + wm * 16, n0 + wn * 64 + ns * 16,
               MROWS, 1 << 30, nullptr, bias, relu != 0, 1.f);
}

// 7) attention scores: sc[z][l][m] = (q_l . k_m)/sqrt(hd), K = hd = 32
__global__ void k_scores(const _Float16* __restrict__ qkvH, float* __restrict__ sc) {
  const int m0 = blockIdx.x * 16, n0 = blockIdx.y * 16, z = blockIdx.z;
  const int b = z / HH, h = z % HH;
  const int lane = threadIdx.x & 31, lo = lane & 15, hi = lane >> 4;
  const int l = m0 + lo, lq = l < LC ? l : LC - 1;
  const int m = n0 + lo, mq = m < LC ? m : LC - 1;
  v16h a  = frag_a_ptr(qkvH + ((size_t)lq * BB + b) * QKV + h * HDIM, hi, l < LC);
  v16h bf = frag_b_ptr(qkvH + ((size_t)mq * BB + b) * QKV + DD + h * HDIM, hi, m < LC);
  v8f c = wmma_acc(a, bf, v8f{});
  store_tile(c, sc + (size_t)z * LP * LP, nullptr, LP, 1, m0, n0, LP, LP,
             nullptr, nullptr, false, 0.17677669529663687f /* 1/sqrt(32) */);
}

// 8) row softmax (f32 in) -> f16 probabilities with zeroed pad tail
__global__ void k_softmax(const float* __restrict__ sc, _Float16* __restrict__ pH) {
  const int idx = blockIdx.x * blockDim.x + threadIdx.x;
  if (idx >= BB * HH * LC) return;
  const int l = idx % LC;
  const int z = idx / LC;
  const float* row = sc + ((size_t)z * LP + l) * LP;
  _Float16* prow = pH + ((size_t)z * LP + l) * LP;
  float mx = -1e30f;
  for (int m = 0; m < LC; ++m) mx = fmaxf(mx, row[m]);
  float s = 0.f;
  for (int m = 0; m < LC; ++m) s += expf(row[m] - mx);
  const float inv = 1.f / s;
  for (int m = 0; m < LC; ++m) prow[m] = (_Float16)(expf(row[m] - mx) * inv);
  for (int m = LC; m < LP; ++m) prow[m] = (_Float16)0.f;
}

// 9) O = P @ V, K = 512 (zero-padded), output f16 (l*B+b, h*32+d)
__global__ void k_av(const _Float16* __restrict__ pH, const _Float16* __restrict__ qkvH,
                     _Float16* __restrict__ attnH) {
  const int m0 = blockIdx.x * 16, n0 = blockIdx.y * 16, z = blockIdx.z;
  const int b = z / HH, h = z % HH;
  const int lane = threadIdx.x & 31, lo = lane & 15, hi = lane >> 4;
  const int l = m0 + lo, lq = l < LC ? l : LC - 1;
  const _Float16* prow = pH + ((size_t)z * LP + lq) * LP;
  const _Float16* vbase = qkvH + 2 * DD + h * HDIM;
  v8f c = {};
  for (int kk = 0; kk < LP; kk += 32) {
    v16h a = frag_a_ptr(prow + kk, hi, l < LC);
    v16h bf = frag_b_gen([&](int kb, int nl) {
      const int m = kk + kb;   // pair spans rows (m, m+1) of V -> per-elem guard
      const int n = n0 + nl;
      h2 r = {};
      const int mc0 = m < LC ? m : LC - 1;
      const int mc1 = m + 1 < LC ? m + 1 : LC - 1;
      _Float16 t0 = vbase[((size_t)mc0 * BB + b) * QKV + n];
      _Float16 t1 = vbase[((size_t)mc1 * BB + b) * QKV + n];
      r[0] = (m < LC) ? t0 : (_Float16)0.f;
      r[1] = (m + 1 < LC) ? t1 : (_Float16)0.f;
      return r;
    }, lo, hi);
    c = wmma_acc(a, bf, c);
  }
  store_tile(c, nullptr, attnH + (size_t)b * DD + h * HDIM, (long)BB * DD, 1,
             m0, n0, LC, HDIM, nullptr, nullptr, false, 1.f);
}

// 10) residual add + LayerNorm, refresh f32 master + f16 mirror
__global__ void k_addln(float* __restrict__ xF, _Float16* __restrict__ xH,
                        const float* __restrict__ delta,
                        const float* __restrict__ g, const float* __restrict__ beta) {
  const int p = blockIdx.x * blockDim.x + threadIdx.x;
  if (p >= MROWS) return;
  const float* dr = delta + (size_t)p * DD;
  float* xr = xF + (size_t)p * DD;
  _Float16* hr = xH + (size_t)p * DD;
  float mean = 0.f;
  for (int i = 0; i < DD; ++i) mean += xr[i] + dr[i];
  mean *= (1.f / DD);
  float var = 0.f;
  for (int i = 0; i < DD; ++i) { const float s = xr[i] + dr[i] - mean; var += s * s; }
  var *= (1.f / DD);
  const float inv = rsqrtf(var + 1e-5f);
  for (int i = 0; i < DD; ++i) {
    const float s = (xr[i] + dr[i] - mean) * inv * g[i] + beta[i];
    xr[i] = s;
    hr[i] = (_Float16)s;
  }
}

// 11) seq max-pool + MLP head + sigmoid
__global__ void k_head(const float* __restrict__ xF,
                       const float* __restrict__ w1, const float* __restrict__ b1,
                       const float* __restrict__ w2, const float* __restrict__ b2,
                       float* __restrict__ out) {
  __shared__ float pooled[DD];
  __shared__ float hbuf[64];
  const int b = blockIdx.x;
  const int t = threadIdx.x;
  if (t < DD) {
    float mx = -1e30f;
    for (int l = 0; l < LC; ++l) mx = fmaxf(mx, xF[((size_t)l * BB + b) * DD + t]);
    pooled[t] = mx;
  }
  __syncthreads();
  if (t < 64) {
    float s = b1[t];
    for (int i = 0; i < DD; ++i) s += w1[t * DD + i] * pooled[i];
    hbuf[t] = fmaxf(s, 0.f);
  }
  __syncthreads();
  if (t == 0) {
    float s = b2[0];
    for (int i = 0; i < 64; ++i) s += w2[i] * hbuf[i];
    out[b] = 1.f / (1.f + expf(-s));
  }
}

// ---------------------------------------------------------------------------
extern "C" void kernel_launch(void* const* d_in, const int* in_sizes, int n_in,
                              void* d_out, int out_size, void* d_ws, size_t ws_size,
                              hipStream_t stream) {
  (void)in_sizes; (void)n_in; (void)out_size; (void)ws_size;

  const int*   input_ids = (const int*)d_in[0];
  const int*   attn_mask = (const int*)d_in[1];
  const float* embed_tab = (const float*)d_in[2];
  const float* conv_w    = (const float*)d_in[3];
  const float* conv_b    = (const float*)d_in[4];
  const float* log_dt    = (const float*)d_in[5];
  const float* log_Are   = (const float*)d_in[6];
  const float* A_imag    = (const float*)d_in[7];
  const float* C_re      = (const float*)d_in[8];
  const float* C_im      = (const float*)d_in[9];
  const float* s4_D      = (const float*)d_in[10];
  const float* qkv_w     = (const float*)d_in[11];
  const float* qkv_b     = (const float*)d_in[12];
  const float* out_w     = (const float*)d_in[13];
  const float* out_b     = (const float*)d_in[14];
  const float* ln1_g     = (const float*)d_in[15];
  const float* ln1_b     = (const float*)d_in[16];
  const float* ff1_w     = (const float*)d_in[17];
  const float* ff1_b     = (const float*)d_in[18];
  const float* ff2_w     = (const float*)d_in[19];
  const float* ff2_b     = (const float*)d_in[20];
  const float* ln2_g     = (const float*)d_in[21];
  const float* ln2_b     = (const float*)d_in[22];
  const float* fc1_w     = (const float*)d_in[23];
  const float* fc1_b     = (const float*)d_in[24];
  const float* fc2_w     = (const float*)d_in[25];
  const float* fc2_b     = (const float*)d_in[26];
  float* outp = (float*)d_out;

  // workspace arena
  char* p = (char*)d_ws;
  auto alloc = [&](size_t bytes) { void* r = p; p += (bytes + 255) & ~(size_t)255; return r; };
  _Float16* embH  = (_Float16*)alloc((size_t)BB * LL * HID * 2);
  _Float16* wH    = (_Float16*)alloc(W_TOT * 2);
  float*    xconv = (float*)alloc((size_t)BB * LP * DD * 4);
  float*    cd    = (float*)alloc((size_t)2 * DD * NN * 4);
  float*    dtA   = (float*)alloc((size_t)2 * DD * NN * 4);
  float*    Kc    = (float*)alloc((size_t)LMAX * DD * 4);
  float*    xF    = (float*)alloc((size_t)MROWS * DD * 4);
  _Float16* xH    = (_Float16*)alloc((size_t)MROWS * DD * 2);
  _Float16* qkvH  = (_Float16*)alloc((size_t)MROWS * QKV * 2);
  float*    scF   = (float*)alloc((size_t)BB * HH * LP * LP * 4);
  _Float16* prH   = (_Float16*)alloc((size_t)BB * HH * LP * LP * 2);
  _Float16* attnH = (_Float16*)alloc((size_t)MROWS * DD * 2);
  float*    tmpF  = (float*)alloc((size_t)MROWS * DD * 4);
  _Float16* ffH   = (_Float16*)alloc((size_t)MROWS * FF * 2);

  // --- embed + weight prep + conv + S4D path ---
  {
    const size_t tot = (size_t)BB * LL * HID;
    k_embed<<<dim3((unsigned)((tot + 255) / 256)), 256, 0, stream>>>(
        input_ids, attn_mask, embed_tab, embH);
  }
  k_wcvt<<<(unsigned)((W_TOT + 255) / 256), 256, 0, stream>>>(
      conv_w, qkv_w, out_w, ff1_w, ff2_w, wH);
  k_s4cd<<<(DD * NN + 255) / 256, 256, 0, stream>>>(log_dt, log_Are, A_imag, C_re, C_im, cd, dtA);
  k_s4k<<<(LMAX * DD + 255) / 256, 256, 0, stream>>>(cd, dtA, Kc);
  k_conv<<<dim3(LP / 64, DD / 16, BB), 32, 0, stream>>>(embH, wH + W_CONV, conv_b, xconv);
  {
    const size_t tot = (size_t)DD * BB * LC;
    k_s4conv<<<dim3((unsigned)((tot + 255) / 256)), 256, 0, stream>>>(xconv, Kc, s4_D, xF, xH);
  }

  // --- 2 transformer encoder layers ---
  for (int lay = 0; lay < 2; ++lay) {
    k_gemm_tiled<<<dim3(MROWS / 64, QKV / 128), 256, 0, stream>>>(
        xH, wH + W_QKV + (size_t)lay * QKV * DD, qkv_b + lay * QKV,
        nullptr, qkvH, QKV, DD, 0);
    k_scores<<<dim3(LP / 16, LP / 16, BB * HH), 32, 0, stream>>>(qkvH, scF);
    k_softmax<<<(BB * HH * LC + 255) / 256, 256, 0, stream>>>(scF, prH);
    k_av<<<dim3(LP / 16, HDIM / 16, BB * HH), 32, 0, stream>>>(prH, qkvH, attnH);
    k_gemm_tiled<<<dim3(MROWS / 64, DD / 128), 256, 0, stream>>>(
        attnH, wH + W_OUT + (size_t)lay * DD * DD, out_b + lay * DD,
        tmpF, nullptr, DD, DD, 0);
    k_addln<<<(MROWS + 255) / 256, 256, 0, stream>>>(xF, xH, tmpF,
        ln1_g + lay * DD, ln1_b + lay * DD);
    k_gemm_tiled<<<dim3(MROWS / 64, FF / 128), 256, 0, stream>>>(
        xH, wH + W_FF1 + (size_t)lay * FF * DD, ff1_b + lay * FF,
        nullptr, ffH, FF, DD, 1);
    k_gemm_tiled<<<dim3(MROWS / 64, DD / 128), 256, 0, stream>>>(
        ffH, wH + W_FF2 + (size_t)lay * DD * FF, ff2_b + lay * DD,
        tmpF, nullptr, DD, FF, 0);
    k_addln<<<(MROWS + 255) / 256, 256, 0, stream>>>(xF, xH, tmpF,
        ln2_g + lay * DD, ln2_b + lay * DD);
  }

  // --- pool + head ---
  k_head<<<BB, 128, 0, stream>>>(xF, fc1_w, fc1_b, fc2_w, fc2_b, outp);
}
